// SelfBlock_32615981645969
// MI455X (gfx1250) — matrix operand order repaired and checked by reading, not compiled, not run on hardware
//
#include <hip/hip_runtime.h>

// ---------------------------------------------------------------------------
// CDNA5 (gfx1250) transformer block: talking-heads attention + ReLU-gated FFN.
// GEMMs: v_wmma_f32_16x16x32_f16 (wave32), 128x64 block tiles, 32x32/wave.
// A-tiles staged with global_load_async_to_lds_b128 (ASYNCcnt), B-tiles
// transposed via LDS scatter. global_prefetch_b8 on streamed operands.
// ---------------------------------------------------------------------------

typedef _Float16 half_t;
typedef __attribute__((ext_vector_type(16))) _Float16 v16h;
typedef __attribute__((ext_vector_type(8)))  float    v8f;

#define DIMC   1024
#define HEADS  16
#define HD     64
#define HIDDEN 2730
#define HPAD   2752      // 43 * 64, 16B-aligned row stride
#define BATCH  4
#define SEQ    1024
#define MTOT   (BATCH * SEQ)   // 4096

struct U4x2 { uint4 a, b; };
struct H8   { half_t h[8]; };

__device__ __forceinline__ v8f wmma_f16(v16h a, v16h b, v8f c) {
  return __builtin_amdgcn_wmma_f32_16x16x32_f16(
      /*neg_a=*/false, a, /*neg_b=*/false, b,
      /*c_mod=*/(short)0, c, /*reuse_a=*/false, /*reuse_b=*/false);
}

// ---- CDNA5 async global->LDS copy (16B), tracked by ASYNCcnt ---------------
__device__ __forceinline__ void async_copy_b128(void* lds_dst, const void* gsrc) {
  const unsigned l = (unsigned)(size_t)lds_dst;        // LDS byte offset
  const unsigned long long g = (unsigned long long)(size_t)gsrc;
  asm volatile("global_load_async_to_lds_b128 %0, %1, off"
               :: "v"(l), "v"(g) : "memory");
}
__device__ __forceinline__ void wait_async() {
  asm volatile("s_wait_asynccnt 0x0" ::: "memory");
}

// ---- A fragment (16x32 f16, row-striped per lane) from LDS -----------------
__device__ __forceinline__ v16h frag_a(const half_t* base) {
  const int lane = threadIdx.x & 31;
  const int row  = lane & 15;
  const int kh   = (lane >> 4) << 3;              // K half-offset: 0 or 8
  const half_t* p = base + row * 32 + kh;
  U4x2 v;
  v.a = *reinterpret_cast<const uint4*>(p);        // K = kh..kh+7
  v.b = *reinterpret_cast<const uint4*>(p + 16);   // K = kh+16..kh+23
  return __builtin_bit_cast(v16h, v);
}

// ---- B fragment (32x16 f16) from K-transposed LDS tile BsT[n][k] -----------
__device__ __forceinline__ v16h frag_b(const half_t* base) {
  const int lane = threadIdx.x & 31;
  const int col  = lane & 15;
  const int kb   = (lane >> 4) << 3;              // K half-offset: 0 or 8
  const half_t* p = base + col * 32 + kb;
  U4x2 v;
  v.a = *reinterpret_cast<const uint4*>(p);        // K = kb..kb+7
  v.b = *reinterpret_cast<const uint4*>(p + 16);   // K = kb+16..kb+23
  return __builtin_bit_cast(v16h, v);
}

// ---- 128x32 A tile: async global->LDS, 256 threads x 16B -------------------
__device__ __forceinline__ void load_a_tile_async(half_t As[128][32],
                                                  const half_t* A, int lda,
                                                  int m0, int k0) {
  const int t  = threadIdx.x;
  const int r  = t >> 1;                 // 0..127
  const int cg = (t & 1) << 4;           // 0 or 16 halves
  async_copy_b128(&As[r][cg], A + (size_t)(m0 + r) * lda + k0 + cg);
}

// ---- 32x64 B tile (row-major [KxN]) -> LDS transposed BsT[n][k] ------------
__device__ __forceinline__ void load_b_n(half_t BsT[64][32], const half_t* Bm,
                                         int ldb, int k0, int n0) {
  const int t  = threadIdx.x;
  const int kr = t >> 3;                 // 0..31
  const int ng = (t & 7) << 3;           // 0..56
  const uint4 v = *reinterpret_cast<const uint4*>(
      Bm + (size_t)(k0 + kr) * ldb + n0 + ng);
  const H8 hv = __builtin_bit_cast(H8, v);
#pragma unroll
  for (int i = 0; i < 8; ++i) BsT[ng + i][kr] = hv.h[i];
}

// ---- B^T tile from [N x K] row-major source: verbatim copy -> async --------
__device__ __forceinline__ void load_b_t_async(half_t BsT[64][32],
                                               const half_t* Km, int ldk,
                                               int k0, int n0) {
  const int t  = threadIdx.x;
  const int nr = t >> 2;                 // 0..63
  const int kg = (t & 3) << 3;           // 0,8,16,24
  async_copy_b128(&BsT[nr][kg], Km + (size_t)(n0 + nr) * ldk + k0 + kg);
}

// ---- 32x32 wave-tile MMA step (4 WMMAs, 2-way A/B fragment reuse) ----------
struct Acc4 { v8f c00, c01, c10, c11; };

__device__ __forceinline__ void mma_step(const half_t (*As)[32],
                                         const half_t (*BsT)[32],
                                         int tm, int tn, Acc4& acc) {
  v16h a0 = frag_a(&As[tm][0]);
  v16h a1 = frag_a(&As[tm + 16][0]);
  v16h b0 = frag_b(&BsT[tn][0]);
  v16h b1 = frag_b(&BsT[tn + 16][0]);
  acc.c00 = wmma_f16(a0, b0, acc.c00);
  acc.c01 = wmma_f16(a0, b1, acc.c01);
  acc.c10 = wmma_f16(a1, b0, acc.c10);
  acc.c11 = wmma_f16(a1, b1, acc.c11);
}

// ===========================================================================
// Elementwise converts
// ===========================================================================
__global__ void cvt_f16_kernel(const float* __restrict__ in,
                               half_t* __restrict__ out, int n) {
  int i = blockIdx.x * 256 + threadIdx.x;
  if (i < n) out[i] = (half_t)in[i];
}

__global__ void cvt_pad_f16_kernel(const float* __restrict__ in,
                                   half_t* __restrict__ out,
                                   int rows, int cols, int ldo, int rows_out) {
  int idx = blockIdx.x * 256 + threadIdx.x;
  int total = rows_out * ldo;
  if (idx >= total) return;
  int r = idx / ldo, c = idx - r * ldo;
  out[idx] = (r < rows && c < cols) ? (half_t)in[(size_t)r * cols + c]
                                    : (half_t)0.f;
}

// ===========================================================================
// LayerNorm: one block per row of 1024, output f16
// ===========================================================================
__global__ void layernorm_kernel(const float* __restrict__ in,
                                 const float* __restrict__ g,
                                 const float* __restrict__ beta,
                                 half_t* __restrict__ out) {
  __shared__ float red[256];
  const int row = blockIdx.x;
  const int t   = threadIdx.x;
  const float* xr = in + (size_t)row * DIMC;
  float v0[4]; float s = 0.f;
#pragma unroll
  for (int i = 0; i < 4; ++i) { v0[i] = xr[t + i * 256]; s += v0[i]; }
  red[t] = s; __syncthreads();
  for (int k = 128; k > 0; k >>= 1) { if (t < k) red[t] += red[t + k]; __syncthreads(); }
  const float mu = red[0] * (1.f / DIMC);
  __syncthreads();
  float var = 0.f;
#pragma unroll
  for (int i = 0; i < 4; ++i) { float d = v0[i] - mu; var += d * d; }
  red[t] = var; __syncthreads();
  for (int k = 128; k > 0; k >>= 1) { if (t < k) red[t] += red[t + k]; __syncthreads(); }
  const float rstd = rsqrtf(red[0] * (1.f / DIMC) + 1e-6f);
#pragma unroll
  for (int i = 0; i < 4; ++i) {
    int c = t + i * 256;
    out[(size_t)row * DIMC + c] = (half_t)((v0[i] - mu) * rstd * g[c] + beta[c]);
  }
}

// ===========================================================================
// QKV GEMM: [4096x1024] @ [1024x3072]; scatter into per-head q(*scale)/k/v
// ===========================================================================
__device__ __forceinline__ void scatter_qkv(int m, int c, float raw,
                                            const float* bias, half_t* q,
                                            half_t* k, half_t* v) {
  const float val = raw + bias[c];
  const int bi = m >> 10, ni = m & 1023;
  const int sel = c >> 10, rem = c & 1023, hh = rem >> 6, ii = rem & 63;
  const size_t dst = ((((size_t)bi * HEADS + hh) << 10) + ni) * HD + ii;
  if (sel == 0)      q[dst] = (half_t)(val * 0.125f);   // HEAD_DIM^-0.5
  else if (sel == 1) k[dst] = (half_t)val;
  else               v[dst] = (half_t)val;
}

__global__ void gemm_qkv_kernel(const half_t* __restrict__ A,
                                const half_t* __restrict__ Bm,
                                const float* __restrict__ bias,
                                half_t* __restrict__ q, half_t* __restrict__ k,
                                half_t* __restrict__ v) {
  __shared__ half_t As[128][32];
  __shared__ half_t BsT[64][32];
  const int m0 = blockIdx.y * 128;
  const int n0 = blockIdx.x * 64;
  const int wid = threadIdx.x >> 5;
  const int wm = wid & 3, wn = wid >> 2;
  Acc4 acc = {};
  for (int k0 = 0; k0 < DIMC; k0 += 32) {
    load_a_tile_async(As, A, DIMC, m0, k0);
    load_b_n(BsT, Bm, 3 * DIMC, k0, n0);
    if (k0 + 32 < DIMC) {
      __builtin_prefetch(A + (size_t)(m0 + (threadIdx.x >> 1)) * DIMC + k0 + 32, 0, 1);
      __builtin_prefetch(Bm + (size_t)(k0 + 32 + (threadIdx.x >> 3)) * (3 * DIMC) + n0, 0, 1);
    }
    wait_async();
    __syncthreads();
    mma_step(As, BsT, wm * 32, wn * 32, acc);
    __syncthreads();
  }
  const int lane = threadIdx.x & 31;
  const int col  = lane & 15;
  const int rb   = (lane >> 4) << 3;
#pragma unroll
  for (int i = 0; i < 8; ++i) {
    const int mA = m0 + wm * 32 + rb + i, mB = mA + 16;
    const int cA = n0 + wn * 32 + col,    cB = cA + 16;
    scatter_qkv(mA, cA, acc.c00[i], bias, q, k, v);
    scatter_qkv(mA, cB, acc.c01[i], bias, q, k, v);
    scatter_qkv(mB, cA, acc.c10[i], bias, q, k, v);
    scatter_qkv(mB, cB, acc.c11[i], bias, q, k, v);
  }
}

// ===========================================================================
// qi[b,h] = q[b,h] (1024x64) @ inter[h] (64x64)
// ===========================================================================
__global__ void gemm_qi_kernel(const half_t* __restrict__ q,
                               const half_t* __restrict__ inter,
                               half_t* __restrict__ qi) {
  __shared__ half_t As[128][32];
  __shared__ half_t BsT[64][32];
  const int z  = blockIdx.z;             // b*16 + h
  const int h  = z & (HEADS - 1);
  const int m0 = blockIdx.y * 128;
  const half_t* A  = q + (size_t)z * (SEQ * HD);
  const half_t* Bm = inter + (size_t)h * (HD * HD);
  const int wid = threadIdx.x >> 5;
  const int wm = wid & 3, wn = wid >> 2;
  Acc4 acc = {};
  for (int k0 = 0; k0 < HD; k0 += 32) {
    load_a_tile_async(As, A, HD, m0, k0);
    load_b_n(BsT, Bm, HD, k0, 0);
    wait_async();
    __syncthreads();
    mma_step(As, BsT, wm * 32, wn * 32, acc);
    __syncthreads();
  }
  const int lane = threadIdx.x & 31;
  const int col  = lane & 15;
  const int rb   = (lane >> 4) << 3;
  half_t* O = qi + (size_t)z * (SEQ * HD);
#pragma unroll
  for (int i = 0; i < 8; ++i) {
    const int mA = m0 + wm * 32 + rb + i, mB = mA + 16;
    const int cA = wn * 32 + col,         cB = cA + 16;
    O[(size_t)mA * HD + cA] = (half_t)acc.c00[i];
    O[(size_t)mA * HD + cB] = (half_t)acc.c01[i];
    O[(size_t)mB * HD + cA] = (half_t)acc.c10[i];
    O[(size_t)mB * HD + cB] = (half_t)acc.c11[i];
  }
}

// ===========================================================================
// logits[b,h] = qi[b,h] (1024x64) @ k[b,h]^T (64x1024)  -> f16
// (both tiles are verbatim copies -> fully async-staged)
// ===========================================================================
__global__ void gemm_logits_kernel(const half_t* __restrict__ qi,
                                   const half_t* __restrict__ kmat,
                                   half_t* __restrict__ S) {
  __shared__ half_t As[128][32];
  __shared__ half_t BsT[64][32];
  const int z  = blockIdx.z;
  const int m0 = blockIdx.y * 128;       // query position
  const int n0 = blockIdx.x * 64;        // key position
  const half_t* A  = qi   + (size_t)z * (SEQ * HD);
  const half_t* Km = kmat + (size_t)z * (SEQ * HD);
  const int wid = threadIdx.x >> 5;
  const int wm = wid & 3, wn = wid >> 2;
  Acc4 acc = {};
  for (int k0 = 0; k0 < HD; k0 += 32) {
    load_a_tile_async(As, A, HD, m0, k0);
    load_b_t_async(BsT, Km, HD, k0, n0);
    wait_async();
    __syncthreads();
    mma_step(As, BsT, wm * 32, wn * 32, acc);
    __syncthreads();
  }
  const int lane = threadIdx.x & 31;
  const int col  = lane & 15;
  const int rb   = (lane >> 4) << 3;
  half_t* O = S + (size_t)z * SEQ * SEQ;
#pragma unroll
  for (int i = 0; i < 8; ++i) {
    const int mA = m0 + wm * 32 + rb + i, mB = mA + 16;
    const int cA = n0 + wn * 32 + col,    cB = cA + 16;
    O[(size_t)mA * SEQ + cA] = (half_t)acc.c00[i];
    O[(size_t)mA * SEQ + cB] = (half_t)acc.c01[i];
    O[(size_t)mB * SEQ + cA] = (half_t)acc.c10[i];
    O[(size_t)mB * SEQ + cB] = (half_t)acc.c11[i];
  }
}

// ===========================================================================
// Per (b,n): pre-softmax head-mix (+bias,+mask), softmax over d,
// post-softmax head-mix (+bias). In-place on the f16 attention buffer.
// 16x1024 f32 slab in LDS (~66KB of the WGP's 320KB).
// ===========================================================================
#define TLD 1028
__global__ void mix_softmax_kernel(half_t* __restrict__ S,
                                   const float* __restrict__ pl_w,
                                   const float* __restrict__ pl_b,
                                   const float* __restrict__ pw_w,
                                   const float* __restrict__ pw_b,
                                   const unsigned char* __restrict__ mask) {
  __shared__ float tile[HEADS][TLD];
  __shared__ float red[256];
  __shared__ float plw_s[HEADS][HEADS], pww_s[HEADS][HEADS];
  __shared__ float plb_s[HEADS], pwb_s[HEADS];

  const int bn = blockIdx.x;             // b*1024 + n
  const int b  = bn >> 10;
  const int n  = bn & 1023;
  const int t  = threadIdx.x;

  if (t < 256) { plw_s[t >> 4][t & 15] = pl_w[t]; pww_s[t >> 4][t & 15] = pw_w[t]; }
  if (t < HEADS) { plb_s[t] = pl_b[t]; pwb_s[t] = pw_b[t]; }

  const size_t base = ((size_t)b * HEADS) * SEQ * SEQ + (size_t)n * SEQ;
#pragma unroll
  for (int h = 0; h < HEADS; ++h) {
    const half_t* src = S + base + (size_t)h * SEQ * SEQ;
    for (int d = t; d < SEQ; d += 256) tile[h][d] = (float)src[d];
  }
  __syncthreads();

  // pre-softmax head mixing, column-wise in place
  for (int d = t; d < SEQ; d += 256) {
    const float adder = mask[b * SEQ + d] ? 0.f : -1e9f;
    float vals[HEADS];
#pragma unroll
    for (int h = 0; h < HEADS; ++h) vals[h] = tile[h][d];
#pragma unroll
    for (int g = 0; g < HEADS; ++g) {
      float acc = plb_s[g] + adder;
#pragma unroll
      for (int h = 0; h < HEADS; ++h) acc += vals[h] * plw_s[h][g];
      tile[g][d] = acc;
    }
  }
  __syncthreads();

  // softmax along d for each head row
  for (int g = 0; g < HEADS; ++g) {
    float m = -3.4e38f;
    for (int d = t; d < SEQ; d += 256) m = fmaxf(m, tile[g][d]);
    red[t] = m; __syncthreads();
    for (int s = 128; s > 0; s >>= 1) { if (t < s) red[t] = fmaxf(red[t], red[t + s]); __syncthreads(); }
    const float rm = red[0];
    __syncthreads();
    float ssum = 0.f;
    for (int d = t; d < SEQ; d += 256) {
      const float e = __expf(tile[g][d] - rm);
      tile[g][d] = e; ssum += e;
    }
    red[t] = ssum; __syncthreads();
    for (int s = 128; s > 0; s >>= 1) { if (t < s) red[t] += red[t + s]; __syncthreads(); }
    const float inv = 1.f / red[0];
    __syncthreads();
    for (int d = t; d < SEQ; d += 256) tile[g][d] *= inv;
    __syncthreads();
  }

  // post-softmax head mixing, write back f16
  for (int d = t; d < SEQ; d += 256) {
    float vals[HEADS];
#pragma unroll
    for (int h = 0; h < HEADS; ++h) vals[h] = tile[h][d];
#pragma unroll
    for (int g = 0; g < HEADS; ++g) {
      float acc = pwb_s[g];
#pragma unroll
      for (int h = 0; h < HEADS; ++h) acc += vals[h] * pww_s[h][g];
      S[base + (size_t)g * SEQ * SEQ + d] = (half_t)acc;
    }
  }
}

// ===========================================================================
// o[b,g] = P[b,g] (1024x1024) @ v[b,g] (1024x64); store interleaved [B,N,C]
// ===========================================================================
__global__ void gemm_av_kernel(const half_t* __restrict__ P,
                               const half_t* __restrict__ vmat,
                               half_t* __restrict__ o) {
  __shared__ half_t As[128][32];
  __shared__ half_t BsT[64][32];
  const int z  = blockIdx.z;
  const int m0 = blockIdx.y * 128;
  const half_t* A  = P    + (size_t)z * SEQ * SEQ;
  const half_t* Bm = vmat + (size_t)z * (SEQ * HD);
  const int wid = threadIdx.x >> 5;
  const int wm = wid & 3, wn = wid >> 2;
  Acc4 acc = {};
  for (int k0 = 0; k0 < SEQ; k0 += 32) {
    load_a_tile_async(As, A, SEQ, m0, k0);
    load_b_n(BsT, Bm, HD, k0, 0);
    if (k0 + 32 < SEQ)
      __builtin_prefetch(A + (size_t)(m0 + (threadIdx.x >> 1)) * SEQ + k0 + 32, 0, 1);
    wait_async();
    __syncthreads();
    mma_step(As, BsT, wm * 32, wn * 32, acc);
    __syncthreads();
  }
  const int lane = threadIdx.x & 31;
  const int col  = lane & 15;
  const int rb   = (lane >> 4) << 3;
  const int bi = z >> 4, hh = z & (HEADS - 1);
#pragma unroll
  for (int i = 0; i < 8; ++i) {
    const int mA = m0 + wm * 32 + rb + i, mB = mA + 16;
    const int cA = wn * 32 + col,         cB = cA + 16;
    const size_t rA = ((size_t)bi * SEQ + mA) * DIMC + hh * HD;
    const size_t rB = ((size_t)bi * SEQ + mB) * DIMC + hh * HD;
    o[rA + cA] = (half_t)acc.c00[i];
    o[rA + cB] = (half_t)acc.c01[i];
    o[rB + cA] = (half_t)acc.c10[i];
    o[rB + cB] = (half_t)acc.c11[i];
  }
}

// ===========================================================================
// proj + residual + layerscale: x1 = x + (o @ Wp + pb) * gamma1
// ===========================================================================
__global__ void gemm_proj_kernel(const half_t* __restrict__ A,
                                 const half_t* __restrict__ Bm,
                                 const float* __restrict__ pb,
                                 const float* __restrict__ x,
                                 const float* __restrict__ gamma1,
                                 float* __restrict__ x1) {
  __shared__ half_t As[128][32];
  __shared__ half_t BsT[64][32];
  const int m0 = blockIdx.y * 128;
  const int n0 = blockIdx.x * 64;
  const int wid = threadIdx.x >> 5;
  const int wm = wid & 3, wn = wid >> 2;
  Acc4 acc = {};
  for (int k0 = 0; k0 < DIMC; k0 += 32) {
    load_a_tile_async(As, A, DIMC, m0, k0);
    load_b_n(BsT, Bm, DIMC, k0, n0);
    if (k0 + 32 < DIMC)
      __builtin_prefetch(A + (size_t)(m0 + (threadIdx.x >> 1)) * DIMC + k0 + 32, 0, 1);
    wait_async();
    __syncthreads();
    mma_step(As, BsT, wm * 32, wn * 32, acc);
    __syncthreads();
  }
  const int lane = threadIdx.x & 31;
  const int col  = lane & 15;
  const int rb   = (lane >> 4) << 3;
#pragma unroll
  for (int i = 0; i < 8; ++i) {
    const int mA = m0 + wm * 32 + rb + i, mB = mA + 16;
    const int cA = n0 + wn * 32 + col,    cB = cA + 16;
    const size_t iA0 = (size_t)mA * DIMC + cA, iA1 = (size_t)mA * DIMC + cB;
    const size_t iB0 = (size_t)mB * DIMC + cA, iB1 = (size_t)mB * DIMC + cB;
    x1[iA0] = x[iA0] + (acc.c00[i] + pb[cA]) * gamma1[cA];
    x1[iA1] = x[iA1] + (acc.c01[i] + pb[cB]) * gamma1[cB];
    x1[iB0] = x[iB0] + (acc.c10[i] + pb[cA]) * gamma1[cA];
    x1[iB1] = x[iB1] + (acc.c11[i] + pb[cB]) * gamma1[cB];
  }
}

// ===========================================================================
// Fused FFN up: ffA = relu(h2 @ Wwide) * (h2 @ Wgate), f16, ld=HPAD
// ===========================================================================
__global__ void gemm_ff1_kernel(const half_t* __restrict__ A,
                                const half_t* __restrict__ Bw,
                                const half_t* __restrict__ Bg,
                                half_t* __restrict__ ffA) {
  __shared__ half_t As[128][32];
  __shared__ half_t BsW[64][32];
  __shared__ half_t BsG[64][32];
  const int m0 = blockIdx.y * 128;
  const int n0 = blockIdx.x * 64;
  const int wid = threadIdx.x >> 5;
  const int wm = wid & 3, wn = wid >> 2;
  Acc4 accW = {}, accG = {};
  for (int k0 = 0; k0 < DIMC; k0 += 32) {
    load_a_tile_async(As, A, DIMC, m0, k0);
    load_b_n(BsW, Bw, HPAD, k0, n0);
    load_b_n(BsG, Bg, HPAD, k0, n0);
    if (k0 + 32 < DIMC)
      __builtin_prefetch(A + (size_t)(m0 + (threadIdx.x >> 1)) * DIMC + k0 + 32, 0, 1);
    wait_async();
    __syncthreads();
    mma_step(As, BsW, wm * 32, wn * 32, accW);
    mma_step(As, BsG, wm * 32, wn * 32, accG);
    __syncthreads();
  }
  const int lane = threadIdx.x & 31;
  const int col  = lane & 15;
  const int rb   = (lane >> 4) << 3;
#pragma unroll
  for (int i = 0; i < 8; ++i) {
    const int mA = m0 + wm * 32 + rb + i, mB = mA + 16;
    const int cA = n0 + wn * 32 + col,    cB = cA + 16;
    ffA[(size_t)mA * HPAD + cA] = (half_t)(fmaxf(accW.c00[i], 0.f) * accG.c00[i]);
    ffA[(size_t)mA * HPAD + cB] = (half_t)(fmaxf(accW.c01[i], 0.f) * accG.c01[i]);
    ffA[(size_t)mB * HPAD + cA] = (half_t)(fmaxf(accW.c10[i], 0.f) * accG.c10[i]);
    ffA[(size_t)mB * HPAD + cB] = (half_t)(fmaxf(accW.c11[i], 0.f) * accG.c11[i]);
  }
}

// ===========================================================================
// FFN down + residual: out = x1 + (ffA @ Wout) * gamma2
// ===========================================================================
__global__ void gemm_out_kernel(const half_t* __restrict__ A,
                                const half_t* __restrict__ Bm,
                                const float* __restrict__ x1,
                                const float* __restrict__ gamma2,
                                float* __restrict__ out) {
  __shared__ half_t As[128][32];
  __shared__ half_t BsT[64][32];
  const int m0 = blockIdx.y * 128;
  const int n0 = blockIdx.x * 64;
  const int wid = threadIdx.x >> 5;
  const int wm = wid & 3, wn = wid >> 2;
  Acc4 acc = {};
  for (int k0 = 0; k0 < HPAD; k0 += 32) {
    load_a_tile_async(As, A, HPAD, m0, k0);
    load_b_n(BsT, Bm, DIMC, k0, n0);
    if (k0 + 32 < HPAD)
      __builtin_prefetch(A + (size_t)(m0 + (threadIdx.x >> 1)) * HPAD + k0 + 32, 0, 1);
    wait_async();
    __syncthreads();
    mma_step(As, BsT, wm * 32, wn * 32, acc);
    __syncthreads();
  }
  const int lane = threadIdx.x & 31;
  const int col  = lane & 15;
  const int rb   = (lane >> 4) << 3;
#pragma unroll
  for (int i = 0; i < 8; ++i) {
    const int mA = m0 + wm * 32 + rb + i, mB = mA + 16;
    const int cA = n0 + wn * 32 + col,    cB = cA + 16;
    const size_t iA0 = (size_t)mA * DIMC + cA, iA1 = (size_t)mA * DIMC + cB;
    const size_t iB0 = (size_t)mB * DIMC + cA, iB1 = (size_t)mB * DIMC + cB;
    out[iA0] = x1[iA0] + acc.c00[i] * gamma2[cA];
    out[iA1] = x1[iA1] + acc.c01[i] * gamma2[cB];
    out[iB0] = x1[iB0] + acc.c10[i] * gamma2[cA];
    out[iB1] = x1[iB1] + acc.c11[i] * gamma2[cB];
  }
}

// ===========================================================================
// Host launch
// ===========================================================================
extern "C" void kernel_launch(void* const* d_in, const int* in_sizes, int n_in,
                              void* d_out, int out_size, void* d_ws, size_t ws_size,
                              hipStream_t stream) {
  const float* x       = (const float*)d_in[0];
  const unsigned char* mask = (const unsigned char*)d_in[1];
  const float* ln1_g   = (const float*)d_in[2];
  const float* ln1_b   = (const float*)d_in[3];
  const float* qkv_w   = (const float*)d_in[4];
  const float* qkv_b   = (const float*)d_in[5];
  const float* inter   = (const float*)d_in[6];
  const float* pl_w    = (const float*)d_in[7];
  const float* pl_b    = (const float*)d_in[8];
  const float* pw_w    = (const float*)d_in[9];
  const float* pw_b    = (const float*)d_in[10];
  const float* proj_w  = (const float*)d_in[11];
  const float* proj_b  = (const float*)d_in[12];
  const float* gamma1  = (const float*)d_in[13];
  const float* ln2_g   = (const float*)d_in[14];
  const float* ln2_b   = (const float*)d_in[15];
  const float* w_wide  = (const float*)d_in[16];
  const float* w_gate  = (const float*)d_in[17];
  const float* w_out   = (const float*)d_in[18];
  const float* gamma2  = (const float*)d_in[19];
  float* out = (float*)d_out;

  char* ws = (char*)d_ws;
  size_t off = 0;
  auto take = [&](size_t bytes) {
    char* p = ws + off;
    off += (bytes + 255) & ~(size_t)255;
    return p;
  };
  half_t* wqkv_h  = (half_t*)take((size_t)DIMC * 3 * DIMC * 2);
  half_t* wproj_h = (half_t*)take((size_t)DIMC * DIMC * 2);
  half_t* wwide_h = (half_t*)take((size_t)DIMC * HPAD * 2);
  half_t* wgate_h = (half_t*)take((size_t)DIMC * HPAD * 2);
  half_t* wout_h  = (half_t*)take((size_t)HPAD * DIMC * 2);
  half_t* inter_h = (half_t*)take((size_t)HEADS * HD * HD * 2);
  half_t* hA      = (half_t*)take((size_t)MTOT * DIMC * 2);
  half_t* q_h     = (half_t*)take((size_t)MTOT * DIMC * 2);
  half_t* k_h     = (half_t*)take((size_t)MTOT * DIMC * 2);
  half_t* v_h     = (half_t*)take((size_t)MTOT * DIMC * 2);
  half_t* qi_h    = (half_t*)take((size_t)MTOT * DIMC * 2);
  half_t* S_h     = (half_t*)take((size_t)BATCH * HEADS * SEQ * SEQ * 2);
  half_t* o_h     = (half_t*)take((size_t)MTOT * DIMC * 2);
  float*  x1      = (float*)take((size_t)MTOT * DIMC * 4);
  half_t* h2_h    = (half_t*)take((size_t)MTOT * DIMC * 2);
  half_t* ffA_h   = (half_t*)take((size_t)MTOT * HPAD * 2);

  const dim3 blk(256);
  // --- weight converts (f32 -> f16, with padding where HIDDEN=2730) ---
  cvt_f16_kernel<<<dim3((DIMC * 3 * DIMC + 255) / 256), blk, 0, stream>>>(
      qkv_w, wqkv_h, DIMC * 3 * DIMC);
  cvt_f16_kernel<<<dim3((DIMC * DIMC + 255) / 256), blk, 0, stream>>>(
      proj_w, wproj_h, DIMC * DIMC);
  cvt_f16_kernel<<<dim3((HEADS * HD * HD + 255) / 256), blk, 0, stream>>>(
      inter, inter_h, HEADS * HD * HD);
  cvt_pad_f16_kernel<<<dim3((DIMC * HPAD + 255) / 256), blk, 0, stream>>>(
      w_wide, wwide_h, DIMC, HIDDEN, HPAD, DIMC);
  cvt_pad_f16_kernel<<<dim3((DIMC * HPAD + 255) / 256), blk, 0, stream>>>(
      w_gate, wgate_h, DIMC, HIDDEN, HPAD, DIMC);
  cvt_pad_f16_kernel<<<dim3((HPAD * DIMC + 255) / 256), blk, 0, stream>>>(
      w_out, wout_h, HIDDEN, DIMC, DIMC, HPAD);

  // --- attention ---
  layernorm_kernel<<<dim3(MTOT), blk, 0, stream>>>(x, ln1_g, ln1_b, hA);
  gemm_qkv_kernel<<<dim3(48, 32), blk, 0, stream>>>(hA, wqkv_h, qkv_b, q_h, k_h, v_h);
  gemm_qi_kernel<<<dim3(1, 8, 64), blk, 0, stream>>>(q_h, inter_h, qi_h);
  gemm_logits_kernel<<<dim3(16, 8, 64), blk, 0, stream>>>(qi_h, k_h, S_h);
  mix_softmax_kernel<<<dim3(MTOT), blk, 0, stream>>>(S_h, pl_w, pl_b, pw_w, pw_b, mask);
  gemm_av_kernel<<<dim3(1, 8, 64), blk, 0, stream>>>(S_h, v_h, o_h);
  gemm_proj_kernel<<<dim3(16, 32), blk, 0, stream>>>(o_h, wproj_h, proj_b, x, gamma1, x1);

  // --- FFN ---
  layernorm_kernel<<<dim3(MTOT), blk, 0, stream>>>(x1, ln2_g, ln2_b, h2_h);
  gemm_ff1_kernel<<<dim3(43, 32), blk, 0, stream>>>(h2_h, wwide_h, wgate_h, ffA_h);
  gemm_out_kernel<<<dim3(16, 32), blk, 0, stream>>>(ffA_h, wout_h, x1, gamma2, out);

  (void)in_sizes; (void)n_in; (void)out_size; (void)ws_size;
}
